// XEncoder_86449101734215
// MI455X (gfx1250) — compile-verified, matrix operand order checked by reading, not run on hardware
//
#include <hip/hip_runtime.h>
#include <hip/hip_bf16.h>

typedef __attribute__((ext_vector_type(16))) _Float16 v16h;
typedef __attribute__((ext_vector_type(8)))  _Float16 v8h;
typedef __attribute__((ext_vector_type(8)))  float    v8f;
typedef __attribute__((ext_vector_type(4)))  int      v4i;

// ---- CDNA5 async global->LDS staging (guarded; fallback = reg pipeline) ----
#if __has_builtin(__builtin_amdgcn_global_load_async_to_lds_b128)
#define GEMM_ASYNC 1
typedef __attribute__((address_space(1))) v4i as1_v4i;
typedef __attribute__((address_space(3))) v4i as3_v4i;
#define ASYNC_CP16(src, dst) \
    __builtin_amdgcn_global_load_async_to_lds_b128( \
        (as1_v4i*)(src), (as3_v4i*)(dst), 0, 0)
static __device__ __forceinline__ void wait_async0() {
#if __has_builtin(__builtin_amdgcn_s_wait_asynccnt)
    __builtin_amdgcn_s_wait_asynccnt(0);
#else
    asm volatile("s_wait_asynccnt 0" ::: "memory");
#endif
}
#else
#define GEMM_ASYNC 0
#endif

static __device__ __forceinline__ v16h ld_frag(const _Float16* p) {
    v8h lo = *(const v8h*)p;
    v8h hi = *(const v8h*)(p + 16);
    return __builtin_shufflevector(lo, hi, 0,1,2,3,4,5,6,7,8,9,10,11,12,13,14,15);
}

// ---------------------------------------------------------------------------
// Generic WMMA GEMM: C = act(A @ Bt^T + bias [+ residual])
//   A  : (M, K) f16 row-major        (+ bz*sA)
//   Bt : (N, K) f16 row-major (B^T)  (+ bz*sB)
//   out index = bz*sC + row*ldc + col  (f32 and/or f16)
//   transposed outs: ot[((row>>8)*N + col)*256 + (row&255)]
// Block tile 64x128, 8 waves (wave32); wave = 16(M) x 64(N) = 4 WMMA tiles.
// Double-buffered LDS, one barrier per K-step (32).
// ---------------------------------------------------------------------------
__global__ __launch_bounds__(256) void gemm_wmma_kernel(
    const _Float16* __restrict__ A, const _Float16* __restrict__ Bt,
    const float* __restrict__ bias, const float* __restrict__ residual,
    float* __restrict__ of32, _Float16* __restrict__ of16,
    float* __restrict__ otf32, _Float16* __restrict__ otf16,
    int M, int N, int K, int ldc,
    long sA, long sB, long sC, int act)
{
    __shared__ __attribute__((aligned(16))) _Float16 Alds[2][64 * 40];
    __shared__ __attribute__((aligned(16))) _Float16 Blds[2][128 * 40];

    const int tid   = threadIdx.x;
    const int wid   = tid >> 5;
    const int lane  = tid & 31;
    const int halfw = lane >> 4;   // K-half select per WMMA VGPR layout
    const int idx   = lane & 15;
    const int waveM = wid & 3;     // 4 M-subtiles of 16
    const int waveN = wid >> 2;    // 2 N-subtiles of 64
    const int bm = blockIdx.y * 64;
    const int bn = blockIdx.x * 128;
    const int bz = blockIdx.z;

    const _Float16* Ab = A  + (size_t)bz * sA;
    const _Float16* Bb = Bt + (size_t)bz * sB;

    const int arow = tid >> 2;         // 0..63
    const int chnk = (tid & 3) * 8;    // 0,8,16,24 halves
    const _Float16* gA  = Ab + (size_t)(bm + arow) * K + chnk;
    const _Float16* gB0 = Bb + (size_t)(bn + arow) * K + chnk;
    const _Float16* gB1 = Bb + (size_t)(bn + 64 + arow) * K + chnk;
    const int aoff  = arow * 40 + chnk;
    const int b0off = arow * 40 + chnk;
    const int b1off = (64 + arow) * 40 + chnk;

    v8f c[4];
    #pragma unroll
    for (int j = 0; j < 4; ++j) c[j] = (v8f){0.f,0.f,0.f,0.f,0.f,0.f,0.f,0.f};

    const int ktn = K >> 5;

#if GEMM_ASYNC
    // ---- async double-buffered pipeline ----
    ASYNC_CP16(gA,  &Alds[0][aoff]);
    ASYNC_CP16(gB0, &Blds[0][b0off]);
    ASYNC_CP16(gB1, &Blds[0][b1off]);
    for (int kt = 0; kt < ktn; ++kt) {
        wait_async0();          // own tile-kt transfers landed in LDS
        __syncthreads();        // visible to all waves; prior reads retired
        if (kt + 1 < ktn) {
            const int s = (kt + 1) & 1;
            ASYNC_CP16(gA  + (kt + 1) * 32, &Alds[s][aoff]);
            ASYNC_CP16(gB0 + (kt + 1) * 32, &Blds[s][b0off]);
            ASYNC_CP16(gB1 + (kt + 1) * 32, &Blds[s][b1off]);
        }
        const int s0 = kt & 1;
        v16h af = ld_frag(&Alds[s0][(waveM * 16 + idx) * 40 + halfw * 8]);
        #pragma unroll
        for (int jt = 0; jt < 4; ++jt) {
            v16h bf = ld_frag(&Blds[s0][(waveN * 64 + jt * 16 + idx) * 40 + halfw * 8]);
            c[jt] = __builtin_amdgcn_wmma_f32_16x16x32_f16(
                        false, af, false, bf, (short)0, c[jt], false, false);
        }
    }
#else
    // ---- synchronous double buffer: regs hold tile kt+1 while LDS holds kt ----
    uint4 ra  = *(const uint4*)gA;
    uint4 rb0 = *(const uint4*)gB0;
    uint4 rb1 = *(const uint4*)gB1;
    for (int kt = 0; kt < ktn; ++kt) {
        const int s0 = kt & 1;
        *(uint4*)&Alds[s0][aoff]  = ra;
        *(uint4*)&Blds[s0][b0off] = rb0;
        *(uint4*)&Blds[s0][b1off] = rb1;
        if (kt + 1 < ktn) {
            if (kt + 2 < ktn) {
                __builtin_prefetch(gA  + (kt + 2) * 32, 0, 1);
                __builtin_prefetch(gB0 + (kt + 2) * 32, 0, 1);
                __builtin_prefetch(gB1 + (kt + 2) * 32, 0, 1);
            }
            ra  = *(const uint4*)(gA  + (kt + 1) * 32);
            rb0 = *(const uint4*)(gB0 + (kt + 1) * 32);
            rb1 = *(const uint4*)(gB1 + (kt + 1) * 32);
        }
        __syncthreads();
        v16h af = ld_frag(&Alds[s0][(waveM * 16 + idx) * 40 + halfw * 8]);
        #pragma unroll
        for (int jt = 0; jt < 4; ++jt) {
            v16h bf = ld_frag(&Blds[s0][(waveN * 64 + jt * 16 + idx) * 40 + halfw * 8]);
            c[jt] = __builtin_amdgcn_wmma_f32_16x16x32_f16(
                        false, af, false, bf, (short)0, c[jt], false, false);
        }
    }
#endif

    #pragma unroll
    for (int jt = 0; jt < 4; ++jt) {
        v8f cc = c[jt];
        const int cl = bn + waveN * 64 + jt * 16 + idx;
        const float bv = bias ? bias[cl] : 0.f;
        #pragma unroll
        for (int r = 0; r < 8; ++r) {
            const int rl = bm + waveM * 16 + halfw * 8 + r;
            float v = cc[r] + bv;
            if (residual) v += residual[(size_t)rl * ldc + cl];
            if      (act == 1) v = fmaxf(v, 0.f);
            else if (act == 2) v = v > 0.f ? v : 0.01f * v;
            else if (act == 3) v = 0.5f * v * (1.f + erff(v * 0.70710678118f));
            else if (act == 4) {  // lorentz distance -> thresholded affinity
                float y = fmaxf(v, 1.f);
                float dd = logf(y + sqrtf(fmaxf(y * y - 1.f, 1e-7f)));
                dd = fminf(fmaxf(dd, 1e-6f), 200.f);
                float a = expf(-dd);
                v = a > 0.8f ? a : 0.f;
            }
            const size_t ci = (size_t)bz * sC + (size_t)rl * ldc + cl;
            if (of32) of32[ci] = v;
            if (of16) of16[ci] = (_Float16)v;
            if (otf32) otf32[((size_t)(rl >> 8) * N + cl) * 256 + (rl & 255)] = v;
            if (otf16) otf16[((size_t)(rl >> 8) * N + cl) * 256 + (rl & 255)] = (_Float16)v;
        }
    }
}

// -------------------- small helper kernels --------------------

__global__ void cvt_f32_f16_kernel(const float* __restrict__ s, _Float16* __restrict__ d, int n) {
    int t = blockIdx.x * 256 + threadIdx.x;
    if (t < n) d[t] = (_Float16)s[t];
}

// W (K,N) f32 -> WT (N,K) f16
__global__ void wtrans_kernel(const float* __restrict__ W, _Float16* __restrict__ WT, int Kd, int Nd) {
    int t = blockIdx.x * 256 + threadIdx.x;
    if (t >= Kd * Nd) return;
    int k = t / Nd, n = t - k * Nd;
    WT[(size_t)n * Kd + k] = (_Float16)W[t];
}

__global__ void disadj_kernel(_Float16* __restrict__ dj) {
    int i = blockIdx.x, j = threadIdx.x;
    dj[i * 256 + j] = (_Float16)expf(-fabsf((float)(i - j)) * 0.36787944117144233f);
}

// Banded attention (WIN=9). One wave per (b,i,h); lane owns 4 of 128 dims.
__global__ __launch_bounds__(256) void attn_kernel(
    const _Float16* __restrict__ Q, const _Float16* __restrict__ Km,
    const _Float16* __restrict__ V, const float* __restrict__ wlp,
    const float* __restrict__ blp, _Float16* __restrict__ O)
{
    const int wid = threadIdx.x >> 5, lane = threadIdx.x & 31;
    const int g = blockIdx.x * 8 + wid;           // 0..65535
    const int h = g & 3;
    const int i = (g >> 2) & 255;
    const int b = g >> 10;
    const int row = b * 256 + i;
    const _Float16* qp = Q + (size_t)row * 512 + h * 128 + lane * 4;
    const float q0 = (float)qp[0], q1 = (float)qp[1], q2 = (float)qp[2], q3 = (float)qp[3];
    const float wl = *wlp, bl = *blp;

    float s[9];
    #pragma unroll
    for (int t9 = 0; t9 < 9; ++t9) {
        const int j = i + t9 - 4;
        float acc = -1e30f;
        if (j >= 0 && j < 256) {
            const _Float16* kp = Km + (size_t)(b * 256 + j) * 512 + h * 128 + lane * 4;
            float p = q0 * (float)kp[0] + q1 * (float)kp[1] + q2 * (float)kp[2] + q3 * (float)kp[3];
            for (int m = 16; m; m >>= 1) p += __shfl_xor(p, m, 32);
            const float d2 = (float)((t9 - 4) * (t9 - 4));
            acc = p * 0.03125f + expf(-fabsf(wl * d2 - bl));
        }
        s[t9] = acc;
    }
    float mx = s[0];
    #pragma unroll
    for (int t9 = 1; t9 < 9; ++t9) mx = fmaxf(mx, s[t9]);
    float Z = 0.f, p9[9];
    #pragma unroll
    for (int t9 = 0; t9 < 9; ++t9) { p9[t9] = (s[t9] > -1e29f) ? expf(s[t9] - mx) : 0.f; Z += p9[t9]; }
    const float inv = 1.f / Z;
    float o0 = 0.f, o1 = 0.f, o2 = 0.f, o3 = 0.f;
    #pragma unroll
    for (int t9 = 0; t9 < 9; ++t9) {
        const int j = i + t9 - 4;
        if (j < 0 || j >= 256) continue;
        const float w = p9[t9] * inv;
        const _Float16* vp = V + (size_t)(b * 256 + j) * 512 + h * 128 + lane * 4;
        o0 += w * (float)vp[0]; o1 += w * (float)vp[1];
        o2 += w * (float)vp[2]; o3 += w * (float)vp[3];
    }
    _Float16* op = O + (size_t)row * 512 + h * 128 + lane * 4;
    op[0] = (_Float16)o0; op[1] = (_Float16)o1; op[2] = (_Float16)o2; op[3] = (_Float16)o3;
}

// Hyperbolic lift: xcat=[xr|c_x] (1024) -> P,G (1056 padded f16), H (1024 f16)
__global__ __launch_bounds__(256) void hyper_kernel(
    const float* __restrict__ xr, const float* __restrict__ cx,
    _Float16* __restrict__ P, _Float16* __restrict__ G, _Float16* __restrict__ Hh)
{
    __shared__ float xc[1024];
    __shared__ float red[256];
    const int r = blockIdx.x, t = threadIdx.x;
    const float a0 = xr[(size_t)r * 512 + t];
    const float a1 = xr[(size_t)r * 512 + 256 + t];
    const float a2 = cx[(size_t)r * 512 + t];
    const float a3 = cx[(size_t)r * 512 + 256 + t];
    xc[t] = a0; xc[256 + t] = a1; xc[512 + t] = a2; xc[768 + t] = a3;
    red[t] = a0 * a0 + a1 * a1 + a2 * a2 + a3 * a3;
    __syncthreads();
    for (int s2 = 128; s2; s2 >>= 1) { if (t < s2) red[t] += red[t + s2]; __syncthreads(); }
    float n = fmaxf(sqrtf(red[0]), 1e-7f);
    const float p0 = coshf(n);
    const float sc = sinhf(n) / n;
    const float y0 = fmaxf(p0, 1.f + 1e-7f);
    const float dl = logf(y0 + sqrtf(y0 * y0 - 1.f));
    const float ns = fmaxf(sinhf(n), 1e-7f);
    const float f = dl * sc / ns;
    const size_t pb = (size_t)r * 1056;
    if (t == 0) { P[pb] = (_Float16)p0; G[pb] = (_Float16)(-p0); }
    for (int e = t; e < 1024; e += 256) {
        const float xv = xc[e];
        const _Float16 pv = (_Float16)(sc * xv);
        P[pb + 1 + e] = pv; G[pb + 1 + e] = pv;
        Hh[(size_t)r * 1024 + e] = (_Float16)(f * xv);
    }
    if (t < 31) { P[pb + 1025 + t] = (_Float16)0.f; G[pb + 1025 + t] = (_Float16)0.f; }
}

// Masked row softmax of thresholded affinity -> f16. One wave per (b,i).
__global__ __launch_bounds__(256) void adjh_kernel(
    const float* __restrict__ thr, const int* __restrict__ seq, _Float16* __restrict__ adjh)
{
    const int wid = threadIdx.x >> 5, lane = threadIdx.x & 31;
    const int row = blockIdx.x * 8 + wid;      // 0..16383
    const int b = row >> 8, i = row & 255;
    const int sl = seq[b];
    const float* tp = thr + (size_t)row * 256;
    float x[8]; float m = -1e30f;
    #pragma unroll
    for (int c = 0; c < 8; ++c) {
        const int j = c * 32 + lane;
        const float v = (j < sl) ? tp[j] : -1e30f;
        x[c] = v; m = fmaxf(m, v);
    }
    for (int mm = 16; mm; mm >>= 1) m = fmaxf(m, __shfl_xor(m, mm, 32));
    float Z = 0.f;
    #pragma unroll
    for (int c = 0; c < 8; ++c) { x[c] = (x[c] > -1e29f) ? expf(x[c] - m) : 0.f; Z += x[c]; }
    for (int mm = 16; mm; mm >>= 1) Z += __shfl_xor(Z, mm, 32);
    const float inv = (i < sl) ? 1.f / Z : 0.f;
    _Float16* op = adjh + (size_t)row * 256;
    #pragma unroll
    for (int c = 0; c < 8; ++c) op[c * 32 + lane] = (_Float16)(x[c] * inv);
}

// Layernorm over 512 -> f16
__global__ __launch_bounds__(256) void ln_kernel(
    const float* __restrict__ xcb, const float* __restrict__ g,
    const float* __restrict__ bta, _Float16* __restrict__ xn)
{
    __shared__ float rs[256], rq[256];
    const int r = blockIdx.x, t = threadIdx.x;
    const float v0 = xcb[(size_t)r * 512 + t];
    const float v1 = xcb[(size_t)r * 512 + 256 + t];
    rs[t] = v0 + v1; rq[t] = v0 * v0 + v1 * v1;
    __syncthreads();
    for (int s2 = 128; s2; s2 >>= 1) { if (t < s2) { rs[t] += rs[t + s2]; rq[t] += rq[t + s2]; } __syncthreads(); }
    const float mu = rs[0] * (1.f / 512.f);
    const float var = rq[0] * (1.f / 512.f) - mu * mu;
    const float inv = rsqrtf(var + 1e-5f);
    xn[(size_t)r * 512 + t]       = (_Float16)((v0 - mu) * inv * g[t] + bta[t]);
    xn[(size_t)r * 512 + 256 + t] = (_Float16)((v1 - mu) * inv * g[256 + t] + bta[256 + t]);
}

// Classifier head: sigmoid(xt . Wcls + bcls). One wave per row.
__global__ __launch_bounds__(256) void cls_kernel(
    const float* __restrict__ xt, const float* __restrict__ Wc,
    const float* __restrict__ bc, float* __restrict__ out)
{
    const int wid = threadIdx.x >> 5, lane = threadIdx.x & 31;
    const int row = blockIdx.x * 8 + wid;
    const float* xp = xt + (size_t)row * 512;
    float acc = 0.f;
    #pragma unroll
    for (int c = 0; c < 16; ++c) { const int j = c * 32 + lane; acc += xp[j] * Wc[j]; }
    for (int m = 16; m; m >>= 1) acc += __shfl_xor(acc, m, 32);
    if (lane == 0) out[row] = 1.f / (1.f + expf(-(acc + bc[0])));
}

// ---------------------------------------------------------------------------
extern "C" void kernel_launch(void* const* d_in, const int* in_sizes, int n_in,
                              void* d_out, int out_size, void* d_ws, size_t ws_size,
                              hipStream_t stream) {
    const float* x    = (const float*)d_in[0];
    const float* cx   = (const float*)d_in[1];
    const float* wl   = (const float*)d_in[2];
    const float* bl   = (const float*)d_in[3];
    const float* Wq   = (const float*)d_in[4];  const float* bq  = (const float*)d_in[5];
    const float* Wk   = (const float*)d_in[6];  const float* bk  = (const float*)d_in[7];
    const float* Wv   = (const float*)d_in[8];  const float* bv  = (const float*)d_in[9];
    const float* Wo   = (const float*)d_in[10]; const float* bo  = (const float*)d_in[11];
    const float* Wc1  = (const float*)d_in[12]; const float* bc1 = (const float*)d_in[13];
    const float* Wg1  = (const float*)d_in[14]; const float* bg1 = (const float*)d_in[15];
    const float* Wg2  = (const float*)d_in[16]; const float* bg2 = (const float*)d_in[17];
    const float* lng  = (const float*)d_in[18]; const float* lnb = (const float*)d_in[19];
    const float* Wl1  = (const float*)d_in[20]; const float* bl1 = (const float*)d_in[21];
    const float* Wcls = (const float*)d_in[22]; const float* bcls= (const float*)d_in[23];
    const int*   seq  = (const int*)d_in[24];

    char* ws = (char*)d_ws;
    const size_t MB = 1048576ull;
    _Float16* WqT  = (_Float16*)(ws + 0 * MB);
    _Float16* WkT  = (_Float16*)(ws + 1 * MB);
    _Float16* WvT  = (_Float16*)(ws + 2 * MB);
    _Float16* WoT  = (_Float16*)(ws + 3 * MB);
    _Float16* Wc1T = (_Float16*)(ws + 4 * MB);
    _Float16* Wg1T = (_Float16*)(ws + 5 * MB);
    _Float16* Wg2T = (_Float16*)(ws + 5 * MB + 524288);
    _Float16* Wl1T = (_Float16*)(ws + 6 * MB);
    _Float16* dj   = (_Float16*)(ws + 6 * MB + 524288 + 262144);
    _Float16* xf16 = (_Float16*)(ws + 8 * MB);
    _Float16* Qf   = (_Float16*)(ws + 40 * MB);
    _Float16* Kf   = (_Float16*)(ws + 56 * MB);
    _Float16* Vf   = (_Float16*)(ws + 72 * MB);
    _Float16* Of   = (_Float16*)(ws + 88 * MB);
    _Float16* x2f  = (_Float16*)(ws + 104 * MB);
    float*    xr   = (float*)   (ws + 136 * MB);
    _Float16* Pf   = (_Float16*)(ws + 8 * MB);    // reuses xf16 (dead)
    _Float16* Gf   = (_Float16*)(ws + 42 * MB);   // reuses Q/K (dead)
    _Float16* Hf   = (_Float16*)(ws + 76 * MB);   // reuses V/O (dead)
    float*    thr  = (float*)   (ws + 108 * MB);
    _Float16* adjh = (_Float16*)(ws + 124 * MB);
    _Float16* g1T  = (_Float16*)(ws + 132 * MB);
    _Float16* g2T  = (_Float16*)(ws + 140 * MB);
    float*    xc   = (float*)   (ws + 148 * MB);
    _Float16* xn   = (_Float16*)(ws + 42 * MB);   // reuses Gf (dead)
    float*    xt   = (float*)   (ws + 8 * MB);    // reuses Pf (dead)

    float* out_xe = (float*)d_out;                // (B,T,1) = 16384
    float* out_xt = (float*)d_out + 16384;        // (B,512,T)

    dim3 blk(256);

    // --- weight transposes (f32 (K,N) -> f16 (N,K)) + x conversion + disadj ---
    wtrans_kernel<<<2048, blk, 0, stream>>>(Wq,  WqT,  1024, 512);
    wtrans_kernel<<<2048, blk, 0, stream>>>(Wk,  WkT,  1024, 512);
    wtrans_kernel<<<2048, blk, 0, stream>>>(Wv,  WvT,  1024, 512);
    wtrans_kernel<<<2048, blk, 0, stream>>>(Wo,  WoT,  512, 1024);
    wtrans_kernel<<<2048, blk, 0, stream>>>(Wc1, Wc1T, 1024, 512);
    wtrans_kernel<<<1024, blk, 0, stream>>>(Wg1, Wg1T, 1024, 256);
    wtrans_kernel<<<1024, blk, 0, stream>>>(Wg2, Wg2T, 1024, 256);
    wtrans_kernel<<<1024, blk, 0, stream>>>(Wl1, Wl1T, 512, 512);
    disadj_kernel<<<256, blk, 0, stream>>>(dj);
    cvt_f32_f16_kernel<<<65536, blk, 0, stream>>>(x, xf16, 16777216);

    // --- G1: Q/K/V = x @ W + b  (f16 out) ---
    gemm_wmma_kernel<<<dim3(4, 256, 1), blk, 0, stream>>>(
        xf16, WqT, bq, nullptr, nullptr, Qf, nullptr, nullptr,
        16384, 512, 1024, 512, 0, 0, 0, 0);
    gemm_wmma_kernel<<<dim3(4, 256, 1), blk, 0, stream>>>(
        xf16, WkT, bk, nullptr, nullptr, Kf, nullptr, nullptr,
        16384, 512, 1024, 512, 0, 0, 0, 0);
    gemm_wmma_kernel<<<dim3(4, 256, 1), blk, 0, stream>>>(
        xf16, WvT, bv, nullptr, nullptr, Vf, nullptr, nullptr,
        16384, 512, 1024, 512, 0, 0, 0, 0);

    // --- banded attention -> Of ---
    attn_kernel<<<8192, blk, 0, stream>>>(Qf, Kf, Vf, wl, bl, Of);

    // --- G2: x2 = O @ Wo + bo + x (residual), f16 out ---
    gemm_wmma_kernel<<<dim3(8, 256, 1), blk, 0, stream>>>(
        Of, WoT, bo, x, nullptr, x2f, nullptr, nullptr,
        16384, 1024, 512, 1024, 0, 0, 0, 0);

    // --- G3: xr = relu(x2 @ Wc1 + bc1), f32 out ---
    gemm_wmma_kernel<<<dim3(4, 256, 1), blk, 0, stream>>>(
        x2f, Wc1T, bc1, nullptr, xr, nullptr, nullptr, nullptr,
        16384, 512, 1024, 512, 0, 0, 0, 1);

    // --- hyperbolic lift -> P, G (1056 padded), H ---
    hyper_kernel<<<16384, blk, 0, stream>>>(xr, cx, Pf, Gf, Hf);

    // --- G4 (batched): thr = lorentz-threshold(G @ P^T), K=1056 ---
    gemm_wmma_kernel<<<dim3(2, 4, 64), blk, 0, stream>>>(
        Gf, Pf, nullptr, nullptr, thr, nullptr, nullptr, nullptr,
        256, 256, 1056, 256, 270336, 270336, 65536, 4);

    // --- masked softmax -> adjh f16 ---
    adjh_kernel<<<2048, blk, 0, stream>>>(thr, seq, adjh);

    // --- G5: g1T/g2T = (H @ Wg + bg)^T, f16 transposed out ---
    gemm_wmma_kernel<<<dim3(2, 256, 1), blk, 0, stream>>>(
        Hf, Wg1T, bg1, nullptr, nullptr, nullptr, nullptr, g1T,
        16384, 256, 1024, 256, 0, 0, 0, 0);
    gemm_wmma_kernel<<<dim3(2, 256, 1), blk, 0, stream>>>(
        Hf, Wg2T, bg2, nullptr, nullptr, nullptr, nullptr, g2T,
        16384, 256, 1024, 256, 0, 0, 0, 0);

    // --- G6 (batched): xc[:, :256]  = leaky(adjh @ g1) ---
    gemm_wmma_kernel<<<dim3(2, 4, 64), blk, 0, stream>>>(
        adjh, g1T, nullptr, nullptr, xc, nullptr, nullptr, nullptr,
        256, 256, 256, 512, 65536, 65536, 131072, 2);
    // --- G7 (batched): xc[:, 256:] = leaky(disadj @ g2), disadj shared (sA=0) ---
    gemm_wmma_kernel<<<dim3(2, 4, 64), blk, 0, stream>>>(
        dj, g2T, nullptr, nullptr, xc + 256, nullptr, nullptr, nullptr,
        256, 256, 256, 512, 0, 65536, 131072, 2);

    // --- layernorm -> xn f16 ---
    ln_kernel<<<16384, blk, 0, stream>>>(xc, lng, lnb, xn);

    // --- G8: xt = gelu(xn @ Wl1 + bl1); also writes xt^T into d_out ---
    gemm_wmma_kernel<<<dim3(4, 256, 1), blk, 0, stream>>>(
        xn, Wl1T, bl1, nullptr, xt, nullptr, out_xt, nullptr,
        16384, 512, 512, 512, 0, 0, 0, 3);

    // --- classifier head -> d_out[:16384] ---
    cls_kernel<<<2048, blk, 0, stream>>>(xt, Wcls, bcls, out_xe);
}
